// SelectiveSSM_26044681683346
// MI455X (gfx1250) — compile-verified, matrix operand order checked by reading, not run on hardware
//
#include <hip/hip_runtime.h>
#include <hip/hip_bf16.h>

// ---------------- problem constants (from reference) ----------------
#define SSM_B   2
#define SSM_L   2048
#define SSM_D   1024
#define SSM_N   16
#define SSM_M   (SSM_B * SSM_L)     // 4096 rows for the projections
#define SSM_K   SSM_D               // 1024 reduction dim
#define CHUNK   64
#define NCHUNK  (SSM_L / CHUNK)     // 32
#define SCAN_PF 4                   // scan prefetch distance (timesteps)

typedef __attribute__((ext_vector_type(16))) __bf16 v16bf;
typedef __attribute__((ext_vector_type(8)))  float  v8f;

// ---------------- fp32 -> bf16 (RNE) conversion ----------------
static __device__ __forceinline__ unsigned short f2bf(float f) {
    unsigned u = __float_as_uint(f);
    u += 0x7FFFu + ((u >> 16) & 1u);   // round-to-nearest-even
    return (unsigned short)(u >> 16);
}

__global__ void f32_to_bf16_kernel(const float* __restrict__ in,
                                   unsigned short* __restrict__ out, int n4) {
    int i = blockIdx.x * blockDim.x + threadIdx.x;
    if (i >= n4) return;
    float4 v = ((const float4*)in)[i];
    unsigned lo = (unsigned)f2bf(v.x) | ((unsigned)f2bf(v.y) << 16);
    unsigned hi = (unsigned)f2bf(v.z) | ((unsigned)f2bf(v.w) << 16);
    ((uint2*)out)[i] = make_uint2(lo, hi);
}

// ---------------- bf16 WMMA GEMM: out = act(Abf @ Wt^T + bias) ----------------
// One wave owns a 16M x (16*NT)N tile.  K is swept 32 at a time with
// v_wmma_f32_16x16x32_bf16.  The K loop is software-pipelined: iteration k+1's
// A/B fragments are loaded (global_load_b128 pairs) BEFORE iteration k's WMMAs
// execute, so the pre-WMMA wait is s_wait_loadcnt <in-flight>, not 0 -- the
// fragment-fetch latency overlaps the matrix pipe within a single wave.
template <int NT, bool SOFTPLUS>
__global__ void gemm_bf16_wmma_kernel(const unsigned short* __restrict__ Abf,
                                      const unsigned short* __restrict__ Wt,
                                      const float* __restrict__ bias,
                                      float* __restrict__ out,
                                      int M, int K, int Nout) {
    const int lane  = threadIdx.x & 31;
    const int wave  = threadIdx.x >> 5;
    const int waves = blockDim.x >> 5;
    const int tile  = blockIdx.x * waves + wave;

    const int ntilesN = Nout / (16 * NT);
    const int tm = tile / ntilesN;
    const int tn = tile % ntilesN;
    if (tm * 16 >= M) return;

    const int l15   = lane & 15;
    const int kbase = (lane < 16) ? 0 : 8;   // 16-bit A/B fragment K offset
    const int mrow  = tm * 16 + l15;
    const int ncol0 = tn * 16 * NT;

    const unsigned short* aRow = Abf + (size_t)mrow * K + kbase;
    const unsigned short* bRow[NT];
#pragma unroll
    for (int t = 0; t < NT; ++t)
        bRow[t] = Wt + (size_t)(ncol0 + t * 16 + l15) * K + kbase;

    v8f acc[NT];
#pragma unroll
    for (int t = 0; t < NT; ++t) acc[t] = (v8f){0.f,0.f,0.f,0.f,0.f,0.f,0.f,0.f};

    // ---- prologue: load fragments for k0 = 0 ----
    v16bf aCur, bCur[NT];
    ((uint4*)&aCur)[0] = *(const uint4*)(aRow);
    ((uint4*)&aCur)[1] = *(const uint4*)(aRow + 16);
#pragma unroll
    for (int t = 0; t < NT; ++t) {
        ((uint4*)&bCur[t])[0] = *(const uint4*)(bRow[t]);
        ((uint4*)&bCur[t])[1] = *(const uint4*)(bRow[t] + 16);
    }

    // ---- pipelined main loop: prefetch k0+32, compute k0 ----
    for (int k0 = 0; k0 < K - 32; k0 += 32) {
        const int k1 = k0 + 32;
        v16bf aNxt, bNxt[NT];
        ((uint4*)&aNxt)[0] = *(const uint4*)(aRow + k1);
        ((uint4*)&aNxt)[1] = *(const uint4*)(aRow + k1 + 16);
#pragma unroll
        for (int t = 0; t < NT; ++t) {
            ((uint4*)&bNxt[t])[0] = *(const uint4*)(bRow[t] + k1);
            ((uint4*)&bNxt[t])[1] = *(const uint4*)(bRow[t] + k1 + 16);
        }
#pragma unroll
        for (int t = 0; t < NT; ++t)
            acc[t] = __builtin_amdgcn_wmma_f32_16x16x32_bf16(
                         false, aCur, false, bCur[t], (short)0, acc[t],
                         false, false);
        aCur = aNxt;
#pragma unroll
        for (int t = 0; t < NT; ++t) bCur[t] = bNxt[t];
    }
    // ---- epilogue compute on the last fragments ----
#pragma unroll
    for (int t = 0; t < NT; ++t)
        acc[t] = __builtin_amdgcn_wmma_f32_16x16x32_bf16(
                     false, aCur, false, bCur[t], (short)0, acc[t],
                     false, false);

    // C/D layout: VGPR r -> row (r + 8*(lane>=16)), col = lane&15
    const int rbase = tm * 16 + ((lane >> 4) << 3);
#pragma unroll
    for (int t = 0; t < NT; ++t) {
        const int ncol = ncol0 + t * 16 + l15;
        const float bv = bias[ncol];
#pragma unroll
        for (int r = 0; r < 8; ++r) {
            float v = acc[t][r] + bv;
            if (SOFTPLUS) v = (v > 20.f) ? v : logf(1.f + __expf(v));
            out[(size_t)(rbase + r) * Nout + ncol] = v;
        }
    }
}

// ---------------- chunked linear scan (pass 1): per-chunk P = prod(a), R ----------------
__global__ void scan_chunk_kernel(const float* __restrict__ x,
                                  const float* __restrict__ A,
                                  const float* __restrict__ delta,
                                  const float* __restrict__ Bmat,
                                  float* __restrict__ P,
                                  float* __restrict__ R) {
    const int d = blockIdx.x * blockDim.x + threadIdx.x;
    const int c = blockIdx.y;
    const int b = blockIdx.z;

    float a[SSM_N], p[SSM_N], r[SSM_N];
#pragma unroll
    for (int n = 0; n < SSM_N; ++n) {
        a[n] = A[(size_t)d * SSM_N + n];
        p[n] = 1.f; r[n] = 0.f;
    }

    const int l0 = c * CHUNK;
    for (int l = l0; l < l0 + CHUNK; ++l) {
        const size_t bl = (size_t)b * SSM_L + l;
        // next-cacheline prefetch: delta/x stride is D*4 = 4KB per step
        if (l + SCAN_PF < l0 + CHUNK) {
            __builtin_prefetch(delta + (bl + SCAN_PF) * SSM_D + d, 0, 0);
            __builtin_prefetch(x     + (bl + SCAN_PF) * SSM_D + d, 0, 0);
        }
        const float dl = delta[bl * SSM_D + d];
        const float xv = x[bl * SSM_D + d];
        const float dx = dl * xv;
        const float4* Bp = (const float4*)(Bmat + bl * SSM_N);
        float4 b4[4] = {Bp[0], Bp[1], Bp[2], Bp[3]};
        const float* bb = (const float*)b4;
#pragma unroll
        for (int n = 0; n < SSM_N; ++n) {
            const float ab = __expf(dl * a[n]);
            p[n] *= ab;
            r[n] = ab * r[n] + dx * bb[n];
        }
    }

    const size_t o = (((size_t)b * SSM_D + d) * NCHUNK + c) * SSM_N;
#pragma unroll
    for (int n = 0; n < SSM_N; ++n) { P[o + n] = p[n]; R[o + n] = r[n]; }
}

// ---------------- pass 2: sequential combine of 32 chunk summaries / channel ----------------
__global__ void scan_combine_kernel(const float* __restrict__ P,
                                    const float* __restrict__ R,
                                    float* __restrict__ Hin) {
    const int idx = blockIdx.x * blockDim.x + threadIdx.x;  // b*D + d
    float h[SSM_N];
#pragma unroll
    for (int n = 0; n < SSM_N; ++n) h[n] = 0.f;
    const size_t base = (size_t)idx * NCHUNK * SSM_N;
    for (int c = 0; c < NCHUNK; ++c) {
        const size_t o = base + (size_t)c * SSM_N;
#pragma unroll
        for (int n = 0; n < SSM_N; ++n) Hin[o + n] = h[n];
#pragma unroll
        for (int n = 0; n < SSM_N; ++n) h[n] = P[o + n] * h[n] + R[o + n];
    }
}

// ---------------- pass 3: replay chunk from correct h_in, emit y ----------------
__global__ void scan_emit_kernel(const float* __restrict__ x,
                                 const float* __restrict__ A,
                                 const float* __restrict__ delta,
                                 const float* __restrict__ Bmat,
                                 const float* __restrict__ Cmat,
                                 const float* __restrict__ Hin,
                                 float* __restrict__ y) {
    const int d = blockIdx.x * blockDim.x + threadIdx.x;
    const int c = blockIdx.y;
    const int b = blockIdx.z;

    float a[SSM_N], h[SSM_N];
    const size_t ho = (((size_t)b * SSM_D + d) * NCHUNK + c) * SSM_N;
#pragma unroll
    for (int n = 0; n < SSM_N; ++n) {
        a[n] = A[(size_t)d * SSM_N + n];
        h[n] = Hin[ho + n];
    }

    const int l0 = c * CHUNK;
    for (int l = l0; l < l0 + CHUNK; ++l) {
        const size_t bl = (size_t)b * SSM_L + l;
        if (l + SCAN_PF < l0 + CHUNK) {
            __builtin_prefetch(delta + (bl + SCAN_PF) * SSM_D + d, 0, 0);
            __builtin_prefetch(x     + (bl + SCAN_PF) * SSM_D + d, 0, 0);
        }
        const float dl = delta[bl * SSM_D + d];
        const float xv = x[bl * SSM_D + d];
        const float dx = dl * xv;
        const float4* Bp = (const float4*)(Bmat + bl * SSM_N);
        const float4* Cp = (const float4*)(Cmat + bl * SSM_N);
        float4 b4[4] = {Bp[0], Bp[1], Bp[2], Bp[3]};
        float4 c4[4] = {Cp[0], Cp[1], Cp[2], Cp[3]};
        const float* bb = (const float*)b4;
        const float* cc = (const float*)c4;
        float acc = 0.f;
#pragma unroll
        for (int n = 0; n < SSM_N; ++n) {
            const float ab = __expf(dl * a[n]);
            h[n] = ab * h[n] + dx * bb[n];
            acc += h[n] * cc[n];
        }
        y[bl * SSM_D + d] = acc;
    }
}

// ---------------- host-side launch ----------------
extern "C" void kernel_launch(void* const* d_in, const int* in_sizes, int n_in,
                              void* d_out, int out_size, void* d_ws, size_t ws_size,
                              hipStream_t stream) {
    const float* x  = (const float*)d_in[0];
    const float* A  = (const float*)d_in[1];
    const float* Wd = (const float*)d_in[2];
    const float* bd = (const float*)d_in[3];
    const float* Wb = (const float*)d_in[4];
    const float* bb = (const float*)d_in[5];
    const float* Wc = (const float*)d_in[6];
    const float* bc = (const float*)d_in[7];
    float* y = (float*)d_out;

    char* ws = (char*)d_ws;
    size_t off = 0;
    auto alloc = [&](size_t bytes) -> char* {
        char* p = ws + off;
        off = (off + bytes + 255) & ~(size_t)255;
        return p;
    };
    float*          delta  = (float*)         alloc((size_t)SSM_M * SSM_D * 4);
    float*          Bmat   = (float*)         alloc((size_t)SSM_M * SSM_N * 4);
    float*          Cmat   = (float*)         alloc((size_t)SSM_M * SSM_N * 4);
    unsigned short* x_bf   = (unsigned short*)alloc((size_t)SSM_M * SSM_K * 2);
    unsigned short* Wd_bf  = (unsigned short*)alloc((size_t)SSM_D * SSM_K * 2);
    unsigned short* Wb_bf  = (unsigned short*)alloc((size_t)SSM_N * SSM_K * 2);
    unsigned short* Wc_bf  = (unsigned short*)alloc((size_t)SSM_N * SSM_K * 2);
    float*          Pchunk = (float*)alloc((size_t)SSM_B * SSM_D * NCHUNK * SSM_N * 4);
    float*          Rchunk = (float*)alloc((size_t)SSM_B * SSM_D * NCHUNK * SSM_N * 4);
    float*          Hin    = (float*)alloc((size_t)SSM_B * SSM_D * NCHUNK * SSM_N * 4);

    // 1) fp32 -> bf16 conversions (memory-bound)
    {
        int n4 = SSM_M * SSM_K / 4;
        f32_to_bf16_kernel<<<(n4 + 255) / 256, 256, 0, stream>>>(x, x_bf, n4);
        n4 = SSM_D * SSM_K / 4;
        f32_to_bf16_kernel<<<(n4 + 255) / 256, 256, 0, stream>>>(Wd, Wd_bf, n4);
        n4 = SSM_N * SSM_K / 4;
        f32_to_bf16_kernel<<<(n4 + 255) / 256, 256, 0, stream>>>(Wb, Wb_bf, n4);
        f32_to_bf16_kernel<<<(n4 + 255) / 256, 256, 0, stream>>>(Wc, Wc_bf, n4);
    }

    // 2) delta = softplus(x @ Wd^T + bd): 256 row-tiles x 16 col-groups (NT=4)
    {
        const int tiles = (SSM_M / 16) * (SSM_D / 64);
        gemm_bf16_wmma_kernel<4, true><<<tiles / 4, 128, 0, stream>>>(
            x_bf, Wd_bf, bd, delta, SSM_M, SSM_K, SSM_D);
    }

    // 3) B and C projections (N=16, NT=1)
    {
        const int tiles = SSM_M / 16;
        gemm_bf16_wmma_kernel<1, false><<<tiles / 4, 128, 0, stream>>>(
            x_bf, Wb_bf, bb, Bmat, SSM_M, SSM_K, SSM_N);
        gemm_bf16_wmma_kernel<1, false><<<tiles / 4, 128, 0, stream>>>(
            x_bf, Wc_bf, bc, Cmat, SSM_M, SSM_K, SSM_N);
    }

    // 4) chunked linear scan: pass1 (parallel), pass2 (combine), pass3 (emit)
    {
        dim3 grid(SSM_D / 256, NCHUNK, SSM_B);
        scan_chunk_kernel<<<grid, 256, 0, stream>>>(x, A, delta, Bmat, Pchunk, Rchunk);
        scan_combine_kernel<<<(SSM_B * SSM_D) / 256, 256, 0, stream>>>(Pchunk, Rchunk, Hin);
        scan_emit_kernel<<<grid, 256, 0, stream>>>(x, A, delta, Bmat, Cmat, Hin, y);
    }
}
